// SimpleBiGRU_90357521973669
// MI455X (gfx1250) — matrix operand order repaired and checked by reading, not compiled, not run on hardware
//
#include <hip/hip_runtime.h>
#include <hip/hip_bf16.h>
#include <math.h>

// ---------------- problem constants ----------------
constexpr int kB  = 64;     // batch
constexpr int kT  = 512;    // time
constexpr int kF  = 256;    // input features
constexpr int kH  = 512;    // hidden
constexpr int kG3 = 3 * kH; // 1536 (r|z|n gate columns)

// ---------------- CDNA5 WMMA vector types ----------------
typedef __attribute__((ext_vector_type(16))) __bf16 v16bf;
typedef __attribute__((ext_vector_type(8)))  __bf16 v8bf;
typedef __attribute__((ext_vector_type(8)))  float  v8f;

// ---------------- workspace layout (bytes) ----------------
constexpr size_t OFF_DATA = 0;                                   // bf16 data
constexpr size_t OFF_WI0  = OFF_DATA + (size_t)kB * kT * kF * 2; // packed Wi fwd
constexpr size_t OFF_WI1  = OFF_WI0 + (size_t)kF * kG3 * 2;      // packed Wi bwd
constexpr size_t OFF_WH0  = OFF_WI1 + (size_t)kF * kG3 * 2;      // packed Wh fwd
constexpr size_t OFF_WH1  = OFF_WH0 + (size_t)kH * kG3 * 2;      // packed Wh bwd
constexpr size_t OFF_XG0  = OFF_WH1 + (size_t)kH * kG3 * 2;      // tiled Xg fwd (fp32)
constexpr size_t OFF_XG1  = OFF_XG0 + (size_t)kT * kB * kG3 * 4; // tiled Xg bwd (fp32)
constexpr size_t WS_NEED  = OFF_XG1 + (size_t)kT * kB * kG3 * 4;

// LDS h row stride (bf16 elems): 512 + 8 -> +16B per row => bank-quad stride 4,
// so a 32-lane ds_load_b128 hits the 2-cycle floor (each quad serves 2 lanes).
constexpr int kHStride = kH + 8;

// ---------------- helpers ----------------
__device__ __forceinline__ v16bf make_a16(const __bf16* p) {
    // 16-bit A-operand layout: this lane's 16 values = p[0..7] and p[16..23]
    v8bf lo = *(const v8bf*)(p);
    v8bf hi = *(const v8bf*)(p + 16);
    v16bf a;
#pragma unroll
    for (int i = 0; i < 8; ++i) { a[i] = lo[i]; a[i + 8] = hi[i]; }
    return a;
}
__device__ __forceinline__ v16bf make_b16(const __bf16* p) {
    // packed B layout: this lane's 16 values = p[0..7] and p[16..23]
    v8bf lo = *(const v8bf*)(p);
    v8bf hi = *(const v8bf*)(p + 16);
    v16bf b;
#pragma unroll
    for (int i = 0; i < 8; ++i) { b[i] = lo[i]; b[i + 8] = hi[i]; }
    return b;
}
__device__ __forceinline__ float fsigmoid(float x) {
    return __builtin_amdgcn_rcpf(1.0f + __expf(-x));
}
__device__ __forceinline__ float ftanh(float x) {
    // 1 - 2/(e^{2x}+1): exact limits (+inf -> 1, -inf -> -1)
    float e = __expf(2.0f * x);
    return 1.0f - 2.0f * __builtin_amdgcn_rcpf(e + 1.0f);
}

// ---------------- conversion / packing ----------------
__global__ void k_cvt_bf16(const float* __restrict__ src, __bf16* __restrict__ dst, int n) {
    int i = blockIdx.x * blockDim.x + threadIdx.x;
    if (i < n) dst[i] = (__bf16)src[i];
}

// Pack row-major W[K][N] (fp32) into WMMA B-operand order:
// out[((k/32)*N + n)*32 + (k%32)] -> each lane reads two 8-bf16 runs, 16 apart.
__global__ void k_pack_b(const float* __restrict__ W, __bf16* __restrict__ out, int K, int N) {
    int e = blockIdx.x * blockDim.x + threadIdx.x;
    if (e >= K * N) return;
    int k = e / N, n = e - k * N;
    out[((size_t)(k >> 5) * N + n) * 32 + (k & 31)] = (__bf16)W[e];
}

// ---------------- input projection GEMM (both directions) ----------------
// Xg = data @ Wi + bi, stored gate-tiled: [t][btile][jgroup][gate][row16][col16]
// One wave computes a 16x64 output tile (4 N-tiles sharing the A operand).
__global__ void __launch_bounds__(256)
k_xgemm(const __bf16* __restrict__ data_bf,
        const __bf16* __restrict__ wiF, const __bf16* __restrict__ wiB,
        const float*  __restrict__ biF, const float*  __restrict__ biB,
        float* __restrict__ xgF, float* __restrict__ xgB) {
    int w    = blockIdx.x * (blockDim.x >> 5) + (threadIdx.x >> 5);
    int lane = threadIdx.x & 31;

    int ng  = w % 24;                 // 24 groups of 64 columns -> kG3
    int bt  = (w / 24) & 3;           // 4 batch tiles of 16
    int t   = (w / 96) % kT;
    int dir = w / (96 * kT);

    const __bf16* wi = dir ? wiB : wiF;
    const float*  bi = dir ? biB : biF;
    float*        xg = dir ? xgB : xgF;

    int lane15 = lane & 15;
    int kofs   = (lane < 16) ? 0 : 8;
    int gsel   = (lane < 16) ? 0 : 8;
    int b      = bt * 16 + lane15;
    int bsrc   = dir ? (kB - 1 - b) : b;  // reference flips batch for bwd

    const __bf16* arow = data_bf + ((size_t)bsrc * kT + t) * kF;

    v8f acc[4] = {};
#pragma unroll
    for (int kt = 0; kt < kF / 32; ++kt) {
        v16bf a = make_a16(arow + kt * 32 + kofs);
#pragma unroll
        for (int nt = 0; nt < 4; ++nt) {
            int colbase = ng * 64 + nt * 16;
            const __bf16* bp = wi + ((size_t)(kt * kG3 + colbase + lane15)) * 32 + gsel;
            v16bf bb = make_b16(bp);
            acc[nt] = __builtin_amdgcn_wmma_f32_16x16x32_bf16(
                false, a, false, bb, (short)0, acc[nt], false, false);
        }
    }

    int rowbase = (lane < 16) ? 0 : 8;
#pragma unroll
    for (int nt = 0; nt < 4; ++nt) {
        int colbase = ng * 64 + nt * 16;
        int gate    = colbase >> 9;          // /512
        int jg      = (colbase & 511) >> 4;  // 16-col group within H
        float bv    = bi[colbase + lane15];
        float* dst  = xg + ((((size_t)t * 4 + bt) * 32 + jg) * 3 + gate) * 256;
#pragma unroll
        for (int i = 0; i < 8; ++i)
            dst[(rowbase + i) * 16 + lane15] = acc[nt][i] + bv;
    }
}

// ---------------- full recurrence, ONE launch ----------------
// 8 blocks = 4 batch tiles x 2 dirs. Batch rows are independent, so each block
// runs all 512 timesteps with only workgroup barriers. 32 waves; wave w owns
// hidden columns [16w,16w+16): computes its r/z/n tiles (3 independent WMMA
// chains, K=512), keeps h in registers (C-layout), shares bf16 h via LDS.
__global__ void __launch_bounds__(1024, 1)
k_recur(const __bf16* __restrict__ whF, const __bf16* __restrict__ whB,
        const float*  __restrict__ xgF, const float*  __restrict__ xgB,
        const float*  __restrict__ bhnF, const float* __restrict__ bhnB,
        float* __restrict__ out) {
    __shared__ __bf16 hlds[16 * kHStride];   // bf16 h tile for next step's A

    int dir = blockIdx.x >> 2;
    int bt  = blockIdx.x & 3;
    int m0  = bt * 16;

    int w    = threadIdx.x >> 5;   // wave id = hidden-column group (0..31)
    int lane = threadIdx.x & 31;
    int lane15 = lane & 15;
    int kofs   = (lane < 16) ? 0 : 8;
    int gsel   = (lane < 16) ? 0 : 8;
    int j0     = w * 16;
    int j      = j0 + lane15;      // this lane's hidden column

    const __bf16* wh  = dir ? whB : whF;
    const float*  xg  = dir ? xgB : xgF;
    const float*  bhn = dir ? bhnB : bhnF;
    int dirofs = dir ? kH : 0;

    // step-invariant Wh base pointers for the 3 gate tiles of this lane
    const __bf16* bpR = wh + ((size_t)(0 * kH + j)) * 32 + gsel;
    const __bf16* bpZ = wh + ((size_t)(1 * kH + j)) * 32 + gsel;
    const __bf16* bpN = wh + ((size_t)(2 * kH + j)) * 32 + gsel;
    float bhnv = bhn[j];

    // zero initial hidden state (LDS bf16 copy + register fp32 state)
    for (int i = threadIdx.x; i < 16 * kHStride; i += 1024)
        hlds[i] = (__bf16)0.0f;
    v8f hreg = {};
    __syncthreads();

#pragma unroll 1
    for (int t = 0; t < kT; ++t) {
        // prefetch this step's pre-activations (independent of h -> hides
        // behind the WMMA chain)
        const float* xgt = xg + ((((size_t)t * 4 + bt) * 32 + w) * 3) * 256;
        float xrv[8], xzv[8], xnv[8];
#pragma unroll
        for (int i = 0; i < 8; ++i) {
            int row = (lane < 16) ? i : i + 8;
            int o   = row * 16 + lane15;
            xrv[i] = xgt[o];
            xzv[i] = xgt[256 + o];
            xnv[i] = xgt[512 + o];
        }

        // hg = h @ Wh for this wave's 3 gate tiles (K = 512)
        v8f ar = {}, az = {}, an = {};
#pragma unroll
        for (int kt = 0; kt < kH / 32; ++kt) {
            v16bf a = make_a16(&hlds[lane15 * kHStride + kt * 32 + kofs]);
            size_t ko = (size_t)kt * kG3 * 32;
            ar = __builtin_amdgcn_wmma_f32_16x16x32_bf16(
                false, a, false, make_b16(bpR + ko), (short)0, ar, false, false);
            az = __builtin_amdgcn_wmma_f32_16x16x32_bf16(
                false, a, false, make_b16(bpZ + ko), (short)0, az, false, false);
            an = __builtin_amdgcn_wmma_f32_16x16x32_bf16(
                false, a, false, make_b16(bpN + ko), (short)0, an, false, false);
        }
        __syncthreads();   // all waves done reading hlds for this step

        // gate math + state update + outputs
#pragma unroll
        for (int i = 0; i < 8; ++i) {
            int row = (lane < 16) ? i : i + 8;
            float r  = fsigmoid(xrv[i] + ar[i]);
            float z  = fsigmoid(xzv[i] + az[i]);
            float n  = ftanh(xnv[i] + r * (an[i] + bhnv));
            float hv = (1.0f - z) * n + z * hreg[i];
            hreg[i] = hv;
            hlds[row * kHStride + j] = (__bf16)hv;
            int bg   = m0 + row;
            int brow = dir ? (kB - 1 - bg) : bg;  // undo batch flip for bwd
            out[((size_t)brow * kT + t) * (2 * kH) + dirofs + j] = hv;
        }
        __syncthreads();   // hlds fully updated before next step's GEMM
    }
}

// ---------------- host launcher ----------------
extern "C" void kernel_launch(void* const* d_in, const int* in_sizes, int n_in,
                              void* d_out, int out_size, void* d_ws, size_t ws_size,
                              hipStream_t stream) {
    (void)in_sizes; (void)n_in; (void)out_size;
    if (ws_size < WS_NEED) return;  // not enough scratch to run safely

    const float* data  = (const float*)d_in[0];
    const float* Wi_f  = (const float*)d_in[1];
    const float* bi_f  = (const float*)d_in[2];
    const float* Wh_f  = (const float*)d_in[3];
    const float* bhn_f = (const float*)d_in[4];
    const float* Wi_b  = (const float*)d_in[5];
    const float* bi_b  = (const float*)d_in[6];
    const float* Wh_b  = (const float*)d_in[7];
    const float* bhn_b = (const float*)d_in[8];
    float* out = (float*)d_out;

    char* ws = (char*)d_ws;
    __bf16* data_bf = (__bf16*)(ws + OFF_DATA);
    __bf16* wiF     = (__bf16*)(ws + OFF_WI0);
    __bf16* wiB     = (__bf16*)(ws + OFF_WI1);
    __bf16* whF     = (__bf16*)(ws + OFF_WH0);
    __bf16* whB     = (__bf16*)(ws + OFF_WH1);
    float*  xgF     = (float*)(ws + OFF_XG0);
    float*  xgB     = (float*)(ws + OFF_XG1);

    // 1) one-shot conversion + weight packing (fully parallel)
    {
        int n = kB * kT * kF;
        k_cvt_bf16<<<(n + 255) / 256, 256, 0, stream>>>(data, data_bf, n);
    }
    k_pack_b<<<(kF * kG3 + 255) / 256, 256, 0, stream>>>(Wi_f, wiF, kF, kG3);
    k_pack_b<<<(kF * kG3 + 255) / 256, 256, 0, stream>>>(Wi_b, wiB, kF, kG3);
    k_pack_b<<<(kH * kG3 + 255) / 256, 256, 0, stream>>>(Wh_f, whF, kH, kG3);
    k_pack_b<<<(kH * kG3 + 255) / 256, 256, 0, stream>>>(Wh_b, whB, kH, kG3);

    // 2) big parallel input-projection GEMM, both directions
    //    waves = 2 dirs * 512 t * 4 btiles * 24 ngroups = 98304; 8 waves/block
    k_xgemm<<<12288, 256, 0, stream>>>(data_bf, wiF, wiB, bi_f, bi_b, xgF, xgB);

    // 3) entire 512-step recurrence in ONE launch (batch rows independent:
    //    only workgroup barriers needed). 8 blocks x 32 waves.
    k_recur<<<8, 1024, 0, stream>>>(whF, whB, xgF, xgB, bhn_f, bhn_b, out);
}
